// OuterProductMean_7215545057290
// MI455X (gfx1250) — matrix operand order repaired and checked by reading, not compiled, not run on hardware
//
#include <hip/hip_runtime.h>

// ---------------------------------------------------------------------------
// OuterProductMean (AlphaFold-style), fused, bf16 WMMA on gfx1250 (wave32).
// B=1, M=128, N=256, C=256, MID=32, OUT_C=128
// ---------------------------------------------------------------------------

typedef __attribute__((ext_vector_type(16))) __bf16 bf16x16;
typedef __attribute__((ext_vector_type(8)))  float  f32x8;

union Frag16 {
    bf16x16 v;
    uint4   q[2];
    unsigned u[8];
    unsigned short h[16];
};

__device__ __forceinline__ unsigned short f2bf(float f) {
    unsigned u = __float_as_uint(f);
    u = (u + 0x7FFFu + ((u >> 16) & 1u)) >> 16;   // round-to-nearest-even
    return (unsigned short)u;
}

// ---------------------------------------------------------------------------
// Kernel 0: one-time weight conversion to bf16.
//   owb  [128][1024]  <- ow  (f32 [OUT_C][MID*MID])
//   wcomb[64][256]    <- rows 0..31 = lw, rows 32..63 = rw
// ---------------------------------------------------------------------------
__global__ __launch_bounds__(256) void prep_weights(
    const float* __restrict__ ow, const float* __restrict__ lw,
    const float* __restrict__ rw,
    unsigned short* __restrict__ owb, unsigned short* __restrict__ wcomb)
{
    int t = blockIdx.x * 256 + threadIdx.x;
    if (t < 128 * 1024) owb[t] = f2bf(ow[t]);
    if (t < 64 * 256) {
        int x = t >> 8, c = t & 255;
        float w = (x < 32) ? lw[x * 256 + c] : rw[(x - 32) * 256 + c];
        wcomb[t] = f2bf(w);
    }
}

// ---------------------------------------------------------------------------
// Kernel 1: LayerNorm + left/right projections (WMMA), masked, bf16 output.
//   Block = 256 threads (8 waves), handles 128 rows (row = m*256 + i).
//   Output layouts: leftG[i][x][m], rightG[j][y][m]  (m contiguous, bf16).
// ---------------------------------------------------------------------------
__global__ __launch_bounds__(256) void ln_proj(
    const float* __restrict__ act,  const float* __restrict__ mask,
    const float* __restrict__ ln_g, const float* __restrict__ ln_b,
    const unsigned short* __restrict__ wcomb,
    const float* __restrict__ lb,   const float* __restrict__ rb,
    unsigned short* __restrict__ leftG, unsigned short* __restrict__ rightG)
{
    __shared__ unsigned short xnb[128 * 256];   // 64 KB normalized rows, bf16

    const int tid  = threadIdx.x;
    const int wave = tid >> 5;
    const int lane = tid & 31;
    const int l15  = lane & 15;
    const int hi   = lane >> 4;          // 0 for lanes 0-15, 1 for 16-31
    const int hi8  = hi * 8;
    const int rowBase = blockIdx.x * 128;

    // ---- Phase 1: LayerNorm, one wave per row, 16 rows per wave -----------
    for (int rr = 0; rr < 16; ++rr) {
        const int rloc = wave + rr * 8;
        const int rg   = rowBase + rloc;
        float a[8];
        {
            const float4* ap = reinterpret_cast<const float4*>(
                act + (size_t)rg * 256 + lane * 8);
            float4 v0 = ap[0], v1 = ap[1];
            a[0]=v0.x; a[1]=v0.y; a[2]=v0.z; a[3]=v0.w;
            a[4]=v1.x; a[5]=v1.y; a[6]=v1.z; a[7]=v1.w;
        }
        float s = 0.f, s2 = 0.f;
        #pragma unroll
        for (int k = 0; k < 8; ++k) { s += a[k]; s2 += a[k] * a[k]; }
        #pragma unroll
        for (int m = 16; m >= 1; m >>= 1) {
            s  += __shfl_xor(s,  m, 32);
            s2 += __shfl_xor(s2, m, 32);
        }
        const float mu  = s * (1.f / 256.f);
        const float var = s2 * (1.f / 256.f) - mu * mu;
        const float rs  = rsqrtf(var + 1e-5f);

        float g[8], bb[8];
        {
            const float4* gp = reinterpret_cast<const float4*>(ln_g + lane * 8);
            const float4* bp = reinterpret_cast<const float4*>(ln_b + lane * 8);
            float4 g0 = gp[0], g1 = gp[1], b0 = bp[0], b1 = bp[1];
            g[0]=g0.x; g[1]=g0.y; g[2]=g0.z; g[3]=g0.w;
            g[4]=g1.x; g[5]=g1.y; g[6]=g1.z; g[7]=g1.w;
            bb[0]=b0.x; bb[1]=b0.y; bb[2]=b0.z; bb[3]=b0.w;
            bb[4]=b1.x; bb[5]=b1.y; bb[6]=b1.z; bb[7]=b1.w;
        }
        unsigned pk[4];
        #pragma unroll
        for (int k = 0; k < 4; ++k) {
            unsigned short h0 = f2bf((a[2*k+0] - mu) * rs * g[2*k+0] + bb[2*k+0]);
            unsigned short h1 = f2bf((a[2*k+1] - mu) * rs * g[2*k+1] + bb[2*k+1]);
            pk[k] = (unsigned)h0 | ((unsigned)h1 << 16);
        }
        *reinterpret_cast<uint4*>(&xnb[rloc * 256 + lane * 8]) =
            make_uint4(pk[0], pk[1], pk[2], pk[3]);
    }
    __syncthreads();

    // ---- Phase 2: WMMA projection [128 x 256] @ [256 x 64] ----------------
    // Wave w owns M-tile rows w*16 .. w*16+15; loops its 4 N-tiles of 16.
    const int rowM = wave * 16;
    for (int nt = 0; nt < 4; ++nt) {
        f32x8 acc = {0.f,0.f,0.f,0.f,0.f,0.f,0.f,0.f};
        const int x = nt * 16 + l15;                 // output column (0..63)
        #pragma unroll
        for (int ks = 0; ks < 8; ++ks) {
            Frag16 A, B;
            const int arow = rowM + l15;
            const unsigned short* ap = &xnb[arow * 256 + ks * 32 + hi8];
            A.q[0] = *reinterpret_cast<const uint4*>(ap);
            A.q[1] = *reinterpret_cast<const uint4*>(ap + 16);
            const unsigned* bp = reinterpret_cast<const unsigned*>(
                wcomb + x * 256 + ks * 32 + hi8);
            #pragma unroll
            for (int r = 0; r < 8; ++r) B.u[r] = bp[r < 4 ? r : r + 4];
            acc = __builtin_amdgcn_wmma_f32_16x16x32_bf16(
                false, A.v, false, B.v, (short)0, acc, false, false);
        }
        const float bias = (x < 32) ? lb[x] : rb[x - 32];
        #pragma unroll
        for (int r = 0; r < 8; ++r) {
            const int rloc = rowM + r + hi8;
            const int rg   = rowBase + rloc;
            const float mk = mask[rg];
            const unsigned short bfv = f2bf(mk * (acc[r] + bias));
            const int m = rg >> 8, i = rg & 255;
            if (x < 32)
                leftG [((size_t)(i * 32 + x))       * 128 + m] = bfv;
            else
                rightG[((size_t)(i * 32 + (x - 32))) * 128 + m] = bfv;
        }
    }
}

// ---------------------------------------------------------------------------
// Kernel 2: fused outer-product-mean + output projection.
//   Grid (16,16): block -> (i-tile 16, j-tile 16). 256 threads = 8 waves.
//   Per ii: GEMM1 (32 x 512, K=128, bf16 WMMA) -> LDS slice [jj][x*32+y]
//           GEMM2 (16 x 128, K=1024, bf16 WMMA) -> out, scaled by 1/(norm+eps)
// ---------------------------------------------------------------------------
__global__ __launch_bounds__(256, 1) void opm_main(
    const unsigned short* __restrict__ leftG,
    const unsigned short* __restrict__ rightG,
    const unsigned short* __restrict__ owb,
    const float* __restrict__ mask, const float* __restrict__ ob,
    float* __restrict__ out)
{
    __shared__ unsigned       rightP[64 * 512];   // 128 KB: K-pair packed B
    __shared__ unsigned short sliceA[16 * 1024];  //  32 KB: GEMM2 A operand
    __shared__ float          normInv[256];       //   1 KB

    const int tid  = threadIdx.x;
    const int wave = tid >> 5;
    const int lane = tid & 31;
    const int l15  = lane & 15;
    const int hi   = lane >> 4;
    const int hi8  = hi * 8;
    const int i0 = blockIdx.x * 16;
    const int j0 = blockIdx.y * 16;

    // ---- Stage right j-tile into LDS, packing m-pairs for B fragments -----
    // rightP[p][col]: col = jj*32+y, p = m/2, value = bf16x2 (m=2p, 2p+1)
    for (int it = 0; it < 128; ++it) {
        const int idx = tid + it * 256;       // 32768 u32 values
        const int p = idx >> 9, col = idx & 511;
        const int jj = col >> 5, y = col & 31;
        rightP[idx] = *reinterpret_cast<const unsigned*>(
            rightG + (size_t)((j0 + jj) * 32 + y) * 128 + 2 * p);
    }
    // ---- Normalization denominators: sum_m mask[m][i]*mask[m][j] ----------
    {
        const int ii = tid >> 4, jj = tid & 15;
        const float* mi = mask + (i0 + ii);
        const float* mj = mask + (j0 + jj);
        float s = 0.f;
        for (int m = 0; m < 128; ++m) s += mi[m * 256] * mj[m * 256];
        normInv[tid] = 1.0f / (s + 1e-3f);
    }
    __syncthreads();

    const int o = wave * 16 + l15;          // this wave's output channels
    const float obv = ob[o];

    for (int ii = 0; ii < 16; ++ii) {
        const int i = i0 + ii;

        // ---- GEMM1: slice[x][jj*32+y] = sum_m left[m,i,x]*right[m,j,y] ----
        for (int mt = 0; mt < 2; ++mt) {
            f32x8 acc[4];
            #pragma unroll
            for (int n = 0; n < 4; ++n)
                acc[n] = (f32x8){0.f,0.f,0.f,0.f,0.f,0.f,0.f,0.f};
            const int x = mt * 16 + l15;
            #pragma unroll
            for (int ks = 0; ks < 4; ++ks) {
                Frag16 A;
                const unsigned short* ap =
                    leftG + ((size_t)(i * 32 + x)) * 128 + ks * 32 + hi8;
                A.q[0] = *reinterpret_cast<const uint4*>(ap);
                A.q[1] = *reinterpret_cast<const uint4*>(ap + 16);
                const int pbase = ks * 16 + hi * 4;
                #pragma unroll
                for (int nt = 0; nt < 4; ++nt) {
                    Frag16 B;
                    const int colb = wave * 64 + nt * 16 + l15;
                    #pragma unroll
                    for (int r = 0; r < 8; ++r)
                        B.u[r] = rightP[(pbase + (r < 4 ? r : r + 4)) * 512 + colb];
                    acc[nt] = __builtin_amdgcn_wmma_f32_16x16x32_bf16(
                        false, A.v, false, B.v, (short)0, acc[nt], false, false);
                }
            }
            #pragma unroll
            for (int nt = 0; nt < 4; ++nt) {
                const int col = wave * 64 + nt * 16 + l15;
                const int jj = col >> 5, y = col & 31;
                #pragma unroll
                for (int r = 0; r < 8; ++r) {
                    const int xr = mt * 16 + r + hi8;
                    sliceA[jj * 1024 + xr * 32 + y] = f2bf(acc[nt][r]);
                }
            }
        }
        __syncthreads();

        // ---- GEMM2: out[jj][o] = sum_xy slice[jj][xy] * ow[o][xy] ---------
        f32x8 acc2 = {0.f,0.f,0.f,0.f,0.f,0.f,0.f,0.f};
        #pragma unroll
        for (int ks = 0; ks < 32; ++ks) {
            Frag16 A, B;
            const unsigned short* ap = &sliceA[l15 * 1024 + ks * 32 + hi8];
            A.q[0] = *reinterpret_cast<const uint4*>(ap);
            A.q[1] = *reinterpret_cast<const uint4*>(ap + 16);
            const unsigned* bp = reinterpret_cast<const unsigned*>(
                owb + (size_t)o * 1024 + ks * 32 + hi8);
            if (ks + 1 < 32) __builtin_prefetch(bp + 16, 0, 1);
            #pragma unroll
            for (int r = 0; r < 8; ++r) B.u[r] = bp[r < 4 ? r : r + 4];
            acc2 = __builtin_amdgcn_wmma_f32_16x16x32_bf16(
                false, A.v, false, B.v, (short)0, acc2, false, false);
        }
        #pragma unroll
        for (int r = 0; r < 8; ++r) {
            const int jj = r + hi8;
            const float inv = normInv[ii * 16 + jj];
            out[(((size_t)i * 256) + (j0 + jj)) * 128 + o] = (acc2[r] + obv) * inv;
        }
        __syncthreads();   // protect sliceA before next ii overwrites it
    }
}

// ---------------------------------------------------------------------------
extern "C" void kernel_launch(void* const* d_in, const int* in_sizes, int n_in,
                              void* d_out, int out_size, void* d_ws, size_t ws_size,
                              hipStream_t stream) {
    (void)in_sizes; (void)n_in; (void)out_size; (void)ws_size;
    const float* act  = (const float*)d_in[0];
    const float* mask = (const float*)d_in[1];
    const float* ln_g = (const float*)d_in[2];
    const float* ln_b = (const float*)d_in[3];
    const float* lw   = (const float*)d_in[4];
    const float* lb   = (const float*)d_in[5];
    const float* rw   = (const float*)d_in[6];
    const float* rb   = (const float*)d_in[7];
    const float* ow   = (const float*)d_in[8];
    const float* ob   = (const float*)d_in[9];
    float* out = (float*)d_out;

    char* ws = (char*)d_ws;
    unsigned short* leftG  = (unsigned short*)(ws);                          // 2 MB
    unsigned short* rightG = (unsigned short*)(ws + (size_t)(2u << 20));     // 2 MB
    unsigned short* owb    = (unsigned short*)(ws + (size_t)(4u << 20));     // 256 KB
    unsigned short* wcomb  = (unsigned short*)(ws + (size_t)(4u << 20) + (256u << 10)); // 32 KB

    prep_weights<<<512, 256, 0, stream>>>(ow, lw, rw, owb, wcomb);
    ln_proj<<<256, 256, 0, stream>>>(act, mask, ln_g, ln_b, wcomb, lb, rb,
                                     leftG, rightG);
    opm_main<<<dim3(16, 16), 256, 0, stream>>>(leftG, rightG, owb, mask, ob, out);
}
